// SelfAttentionWithCache_23124103922214
// MI455X (gfx1250) — matrix operand order repaired and checked by reading, not compiled, not run on hardware
//
#include <hip/hip_runtime.h>
#include <hip/hip_bf16.h>

// ---------------- types ----------------
typedef __attribute__((ext_vector_type(16))) __bf16        v16bf;
typedef __attribute__((ext_vector_type(8)))  float         v8f;
typedef __attribute__((ext_vector_type(4)))  unsigned int  u32x4;
typedef __attribute__((ext_vector_type(4)))  int           v4i_;

#define AS1 __attribute__((address_space(1)))
#define AS3 __attribute__((address_space(3)))

union Frag {
  u32x4 q[2];
  v16bf v;
  unsigned short us[16];
};

__device__ __forceinline__ unsigned short f32_to_bf16(float f) {
  union { float f; unsigned int u; } cv;
  cv.f = f;
  unsigned int u = cv.u;
  u += 0x7fffu + ((u >> 16) & 1u);   // round-to-nearest-even
  return (unsigned short)(u >> 16);
}

// constants
#define BATCH   2
#define SLEN    2048
#define DMODEL  2048
#define NHEADS  16
#define HDIM    128
#define MROWS   (BATCH * SLEN)   // 4096
#define KDIM    2048             // GEMM K (compile-time for clean pipelining)

// async-to-LDS availability (CDNA5 / gfx1250)
#if __has_builtin(__builtin_amdgcn_global_load_async_to_lds_b128)
#define HAVE_ASYNC_LDS 1
#else
#define HAVE_ASYNC_LDS 0
#endif

__device__ __forceinline__ void wait_asynccnt0() {
#if __has_builtin(__builtin_amdgcn_s_wait_asynccnt)
  __builtin_amdgcn_s_wait_asynccnt(0);
#else
  asm volatile("s_wait_asynccnt 0x0" ::: "memory");
#endif
}

// ---------------- fp32 -> bf16 convert ----------------
__global__ __launch_bounds__(256)
void cvt_bf16_kernel(const float* __restrict__ src, unsigned short* __restrict__ dst, int n) {
  int i = blockIdx.x * 256 + threadIdx.x;
  if (i < n) dst[i] = f32_to_bf16(src[i]);
}

// ---------------- GEMM: C[m,n] = scale * sum_k A[m,k] * W[n,k]  (y = A @ W^T) ----------
// A: [M,K] bf16 row-major.  W: [N,K] bf16 row-major.  K = N = 2048 fixed.
// 256 threads = 8 waves; wave computes 32(M) x 64(N) with register double-buffering.
// grid = (N/512, M/32).
struct FragSet {
  Frag a[2];   // two 16-row A tiles
  Frag b[4];   // four 16-col B tiles
};

__device__ __forceinline__ void load_frags(FragSet& f,
                                           const unsigned short* ar0,
                                           const unsigned short* ar1,
                                           const unsigned short* const* wr,
                                           int k0, int hh) {
  // A layout (16-bit, 16x32): half h holds K = [h*8, h*8+7] U [16+h*8, 16+h*8+7]
  f.a[0].q[0] = *(const u32x4*)(ar0 + k0 + hh * 8);
  f.a[0].q[1] = *(const u32x4*)(ar0 + k0 + 16 + hh * 8);
  f.a[1].q[0] = *(const u32x4*)(ar1 + k0 + hh * 8);
  f.a[1].q[1] = *(const u32x4*)(ar1 + k0 + 16 + hh * 8);
#pragma unroll
  for (int t = 0; t < 4; ++t) {
    // B operand: half h holds K = [h*8, h*8+7] U [16+h*8, 16+h*8+7] (wr[t] already includes hh*16)
    f.b[t].q[0] = *(const u32x4*)(wr[t] + k0 - hh * 8);
    f.b[t].q[1] = *(const u32x4*)(wr[t] + k0 + 16 - hh * 8);
  }
}

__device__ __forceinline__ void do_wmma8(v8f* acc, const FragSet& f) {
#pragma unroll
  for (int t = 0; t < 4; ++t) {
    acc[t] = __builtin_amdgcn_wmma_f32_16x16x32_bf16(
        false, f.a[0].v, false, f.b[t].v, (short)0, acc[t], false, false);
    acc[4 + t] = __builtin_amdgcn_wmma_f32_16x16x32_bf16(
        false, f.a[1].v, false, f.b[t].v, (short)0, acc[4 + t], false, false);
  }
}

__global__ __launch_bounds__(256)
void gemm_bf16_wmma(const unsigned short* __restrict__ A,
                    const unsigned short* __restrict__ W,
                    unsigned short* __restrict__ bdst,
                    float* __restrict__ fdst,
                    float scale, int head_layout) {
  const int lane = threadIdx.x & 31;
  const int wave = threadIdx.x >> 5;
  const int m0 = blockIdx.y * 32;
  const int n0 = blockIdx.x * 512 + wave * 64;
  const int lm = lane & 15;
  const int hh = lane >> 4;

  v8f zero = {};
  v8f acc[8];
#pragma unroll
  for (int t = 0; t < 8; ++t) acc[t] = zero;

  const unsigned short* ar0 = A + (size_t)(m0 + lm) * KDIM;
  const unsigned short* ar1 = ar0 + (size_t)16 * KDIM;
  const unsigned short* wr[4];
#pragma unroll
  for (int t = 0; t < 4; ++t)
    wr[t] = W + (size_t)(n0 + t * 16 + lm) * KDIM + hh * 16;

  // software pipeline: ping-pong fragment sets so loads overlap WMMA
  FragSet fs0, fs1;
  load_frags(fs0, ar0, ar1, wr, 0, hh);
  for (int k0 = 0; k0 < KDIM; k0 += 64) {
    load_frags(fs1, ar0, ar1, wr, k0 + 32, hh);
    do_wmma8(acc, fs0);
    if (k0 + 64 < KDIM) load_frags(fs0, ar0, ar1, wr, k0 + 64, hh);
    do_wmma8(acc, fs1);
  }

  // C/D layout: VGPR r, lanes 0-15 -> M=r ; lanes 16-31 -> M=8+r ; N = lane%16
#pragma unroll
  for (int mi = 0; mi < 2; ++mi) {
#pragma unroll
    for (int t = 0; t < 4; ++t) {
#pragma unroll
      for (int r = 0; r < 8; ++r) {
        int row = m0 + mi * 16 + hh * 8 + r;
        int col = n0 + t * 16 + lm;
        float val = acc[mi * 4 + t][r] * scale;
        if (head_layout) {
          int bb = row >> 11, ss = row & (SLEN - 1);
          int head = col >> 7, hdi = col & (HDIM - 1);
          size_t idx = (((size_t)(bb * NHEADS + head)) * SLEN + ss) * HDIM + hdi;
          if (bdst) bdst[idx] = f32_to_bf16(val);
          if (fdst) fdst[idx] = val;
        } else {
          size_t idx = (size_t)row * DMODEL + col;
          if (fdst) fdst[idx] = val;
          if (bdst) bdst[idx] = f32_to_bf16(val);
        }
      }
    }
  }
}

// ---------------- causal flash attention ----------------
// grid = (S/64, H, B), block = 128 (4 waves). Wave w: 16 queries at qblk*64+w*16.
// q is pre-scaled by 1/sqrt(hd). Key blocks of 32.
#define KB 32
#define VSTR 40   // vT LDS stride (ushorts); 80B rows keep 16B alignment

__global__ __launch_bounds__(128)
void attn_kernel(const unsigned short* __restrict__ qb,
                 const unsigned short* __restrict__ kb,
                 const unsigned short* __restrict__ vb,
                 unsigned short* __restrict__ ab) {
  const int qblk = blockIdx.x, h = blockIdx.y, b = blockIdx.z;
  const int tid  = threadIdx.x;
  const int wave = tid >> 5;
  const int lane = tid & 31;
  const int lm = lane & 15;
  const int hh = lane >> 4;
  const int qbase = qblk * 64 + wave * 16;

  const size_t headoff = (size_t)(b * NHEADS + h) * SLEN * HDIM;
  const unsigned short* qptr = qb + headoff;
  const unsigned short* kptr = kb + headoff;
  const unsigned short* vptr = vb + headoff;

  __shared__ __align__(16) unsigned short ks [KB * HDIM];     // 8 KB
  __shared__ __align__(16) unsigned short vts[HDIM * VSTR];   // 10 KB (transposed V)
  __shared__ __align__(16) unsigned short ps [4][16 * 32];    // 4 KB  (per-wave P staging)

  // preload Q fragments (A operands, 4 K-chunks over hd=128)
  v16bf aq[4];
  {
    const unsigned short* rp = qptr + (size_t)(qbase + lm) * HDIM;
#pragma unroll
    for (int kk = 0; kk < 4; ++kk) {
      Frag a;
      a.q[0] = *(const u32x4*)(rp + kk * 32 + hh * 8);
      a.q[1] = *(const u32x4*)(rp + kk * 32 + 16 + hh * 8);
      aq[kk] = a.v;
    }
  }

  v8f zero = {};
  v8f o[8];
#pragma unroll
  for (int t = 0; t < 8; ++t) o[t] = zero;
  float mrow[8], lrow[8];
#pragma unroll
  for (int r = 0; r < 8; ++r) { mrow[r] = -1e30f; lrow[r] = 0.0f; }

  const int nkb = 2 * qblk + 2;   // causal: keys up to qblk*64+63

  for (int kbI = 0; kbI < nkb; ++kbI) {
    __syncthreads();   // previous iteration consumers done before overwrite
    // --- stage K tile [32 x 128] bf16 into LDS ---
#if HAVE_ASYNC_LDS
#pragma unroll
    for (int i = 0; i < 4; ++i) {
      int chunk = tid * 4 + i;          // 512 chunks of 8 ushorts (16B)
      int kr = chunk >> 4;
      int cc = (chunk & 15) * 8;
      __builtin_amdgcn_global_load_async_to_lds_b128(
          (AS1 v4i_*)(kptr + (size_t)(kbI * KB + kr) * HDIM + cc),
          (AS3 v4i_*)(ks + kr * HDIM + cc),
          0, 0);
    }
#else
#pragma unroll
    for (int i = 0; i < 4; ++i) {
      int chunk = tid * 4 + i;
      int kr = chunk >> 4;
      int cc = (chunk & 15) * 8;
      *(u32x4*)(ks + kr * HDIM + cc) =
          *(const u32x4*)(kptr + (size_t)(kbI * KB + kr) * HDIM + cc);
    }
#endif
    // --- stage V tile transposed -> vts[col][key] ---
    {
      int key = tid >> 2;
      int colbase = (tid & 3) * 32;
      const unsigned short* vrow = vptr + (size_t)(kbI * KB + key) * HDIM + colbase;
#pragma unroll
      for (int j = 0; j < 32; ++j) vts[(colbase + j) * VSTR + key] = vrow[j];
    }
#if HAVE_ASYNC_LDS
    wait_asynccnt0();
#endif
    __syncthreads();

    if (kbI * KB <= qbase + 15) {       // wave-uniform causal skip
      // ---- scores: S = Q @ K^T, two 16x16 tiles (keys 0-15, 16-31) ----
      v8f s0 = zero, s1 = zero;
#pragma unroll
      for (int kk = 0; kk < 4; ++kk) {
        Frag b0, b1;
        const unsigned short* p0 = ks + (size_t)lm * HDIM + kk * 32 + hh * 8;
        const unsigned short* p1 = ks + (size_t)(lm + 16) * HDIM + kk * 32 + hh * 8;
        b0.q[0] = *(const u32x4*)(p0);  b0.q[1] = *(const u32x4*)(p0 + 16);
        b1.q[0] = *(const u32x4*)(p1);  b1.q[1] = *(const u32x4*)(p1 + 16);
        s0 = __builtin_amdgcn_wmma_f32_16x16x32_bf16(false, aq[kk], false, b0.v, (short)0, s0, false, false);
        s1 = __builtin_amdgcn_wmma_f32_16x16x32_bf16(false, aq[kk], false, b1.v, (short)0, s1, false, false);
      }

      // ---- mask + online softmax (row = hh*8+r, col = lane%16 [+16]) ----
      const int c0 = kbI * KB + lm;
      const int c1 = c0 + 16;
      float corr[8];
#pragma unroll
      for (int r = 0; r < 8; ++r) {
        int qi = qbase + hh * 8 + r;
        float x0 = (c0 > qi) ? -1e30f : s0[r];
        float x1 = (c1 > qi) ? -1e30f : s1[r];
        float mx = fmaxf(x0, x1);
#pragma unroll
        for (int off = 1; off < 16; off <<= 1)
          mx = fmaxf(mx, __shfl_xor(mx, off, 32));   // stays within 16-lane half
        float mnew = fmaxf(mrow[r], mx);
        corr[r] = __expf(mrow[r] - mnew);
        float p0 = __expf(x0 - mnew);
        float p1 = __expf(x1 - mnew);
        float rs = p0 + p1;
#pragma unroll
        for (int off = 1; off < 16; off <<= 1)
          rs += __shfl_xor(rs, off, 32);
        lrow[r] = lrow[r] * corr[r] + rs;
        mrow[r] = mnew;
        s0[r] = p0;
        s1[r] = p1;
      }
#pragma unroll
      for (int t = 0; t < 8; ++t)
#pragma unroll
        for (int r = 0; r < 8; ++r) o[t][r] *= corr[r];

      // ---- P: C-layout -> A-layout via per-wave LDS (in-order per wave) ----
#pragma unroll
      for (int r = 0; r < 8; ++r) {
        int row = hh * 8 + r;
        ps[wave][row * 32 + lm]      = f32_to_bf16(s0[r]);
        ps[wave][row * 32 + 16 + lm] = f32_to_bf16(s1[r]);
      }
      Frag ap;
      ap.q[0] = *(const u32x4*)(&ps[wave][lm * 32 + hh * 8]);
      ap.q[1] = *(const u32x4*)(&ps[wave][lm * 32 + 16 + hh * 8]);

      // ---- O += P @ V, 8 hd-column tiles ----
#pragma unroll
      for (int t = 0; t < 8; ++t) {
        const unsigned short* vp = vts + (size_t)(t * 16 + lm) * VSTR + hh * 8;
        Frag bv;
        bv.q[0] = *(const u32x4*)(vp);
        bv.q[1] = *(const u32x4*)(vp + 16);
        o[t] = __builtin_amdgcn_wmma_f32_16x16x32_bf16(false, ap.v, false, bv.v, (short)0, o[t], false, false);
      }
    }
  }

  // ---- finalize: out = O / l, write bf16 [B*S, D] (D index = h*128 + col) ----
#pragma unroll
  for (int r = 0; r < 8; ++r) {
    float inv = 1.0f / lrow[r];
    int qi = qbase + hh * 8 + r;
    size_t rowbase = ((size_t)b * SLEN + qi) * DMODEL + h * HDIM;
#pragma unroll
    for (int t = 0; t < 8; ++t)
      ab[rowbase + t * 16 + lm] = f32_to_bf16(o[t][r] * inv);
  }
}

// ---------------- launcher ----------------
extern "C" void kernel_launch(void* const* d_in, const int* in_sizes, int n_in,
                              void* d_out, int out_size, void* d_ws, size_t ws_size,
                              hipStream_t stream) {
  const float* x  = (const float*)d_in[0];
  const float* Wq = (const float*)d_in[1];
  const float* Wk = (const float*)d_in[2];
  const float* Wv = (const float*)d_in[3];
  const float* Wo = (const float*)d_in[4];
  float* out_f = (float*)d_out;

  const size_t XN = (size_t)MROWS * DMODEL;        // 8,388,608
  const size_t WN = (size_t)DMODEL * DMODEL;       // 4,194,304

  unsigned short* ws = (unsigned short*)d_ws;
  unsigned short* xb  = ws;
  unsigned short* wqb = xb  + XN;
  unsigned short* wkb = wqb + WN;
  unsigned short* wvb = wkb + WN;
  unsigned short* wob = wvb + WN;
  unsigned short* qb  = wob + WN;
  unsigned short* kbf = qb  + XN;
  unsigned short* vbf = kbf + XN;
  unsigned short* ab  = vbf + XN;                  // total ~112 MB

  float* k_out = out_f + XN;       // d_out: [out | k | v]
  float* v_out = out_f + 2 * XN;

  // 1) fp32 -> bf16
  cvt_bf16_kernel<<<(int)(XN / 256), 256, 0, stream>>>(x,  xb,  (int)XN);
  cvt_bf16_kernel<<<(int)(WN / 256), 256, 0, stream>>>(Wq, wqb, (int)WN);
  cvt_bf16_kernel<<<(int)(WN / 256), 256, 0, stream>>>(Wk, wkb, (int)WN);
  cvt_bf16_kernel<<<(int)(WN / 256), 256, 0, stream>>>(Wv, wvb, (int)WN);
  cvt_bf16_kernel<<<(int)(WN / 256), 256, 0, stream>>>(Wo, wob, (int)WN);

  // 2) Q/K/V projections (y = x @ W^T), outputs in [B,H,S,hd] layout
  dim3 ggrid(DMODEL / 512, MROWS / 32);
  const float qscale = 0.08838834764831845f;       // 1/sqrt(128)
  gemm_bf16_wmma<<<ggrid, 256, 0, stream>>>(xb, wqb, qb,  nullptr, qscale, 1);
  gemm_bf16_wmma<<<ggrid, 256, 0, stream>>>(xb, wkb, kbf, k_out,   1.0f,   1);
  gemm_bf16_wmma<<<ggrid, 256, 0, stream>>>(xb, wvb, vbf, v_out,   1.0f,   1);

  // 3) causal attention
  attn_kernel<<<dim3(SLEN / 64, NHEADS, BATCH), 128, 0, stream>>>(qb, kbf, vbf, ab);

  // 4) output projection -> fp32 d_out
  gemm_bf16_wmma<<<ggrid, 256, 0, stream>>>(ab, wob, nullptr, out_f, 1.0f, 0);
}
